// Adaptive_Spatial_Attention_20366734918243
// MI455X (gfx1250) — matrix operand order repaired and checked by reading, not compiled, not run on hardware
//
#include <hip/hip_runtime.h>
#include <hip/hip_bf16.h>
#include <math.h>

// ---------------- problem constants (from setup_inputs) ----------------
#define BSZ   2
#define HH    64
#define WW    64
#define LL    (HH * WW)      // 4096
#define CC    96
#define NN    16
#define KK    4              // scan directions
#define RR    6              // dt rank
#define DBL   38             // R + 2N
#define DBLP  48             // padded to 3 WMMA tiles
#define BKN   (BSZ * KK)     // 8
#define TCH   128            // scan chunk length
#define NCHUNK (LL / TCH)    // 32
#define BNSCALE 0.999995000037f  // 1/sqrt(1+1e-5)

typedef __attribute__((ext_vector_type(16))) _Float16 v16h;
typedef __attribute__((ext_vector_type(8)))  float    v8f;

__device__ __forceinline__ float gelu_exact(float x) {
    return 0.5f * x * (1.0f + erff(x * 0.70710678118654752f));
}
__device__ __forceinline__ float sigmoidf(float x) {
    return 1.0f / (1.0f + __expf(-x));
}
__device__ __forceinline__ float softplusf(float x) {
    return (x > 20.0f) ? x : __logf(1.0f + __expf(x));
}

// ======================================================================
// WMMA GEMM:  OUT(M,C) = X(M,C) @ Wt(C,C)^T (+bias)    f16 in / f32 acc
// ======================================================================
__global__ void k_wmma_gemm_nt(const float* __restrict__ X,
                               const float* __restrict__ Wt,   // (N,K) row-major
                               const float* __restrict__ bias, // may be null
                               float* __restrict__ OUT,
                               int M) {
    const int wave = threadIdx.x >> 5;
    const int lane = threadIdx.x & 31;
    const int tilesN = CC / 16;                       // 6
    const int tile = blockIdx.x * 8 + wave;
    const int tm = (tile / tilesN) * 16;
    const int tn = (tile % tilesN) * 16;
    if (tm >= M) return;

    const int l15   = lane & 15;
    const bool lo   = (lane < 16);
    const int m     = tm + l15;
    const int n     = tn + l15;

    v8f acc = {};
    for (int k0 = 0; k0 < CC; k0 += 32) {
        v16h a, b;
        const float* xr = X + (size_t)m * CC + k0;
        const int kbase = lo ? 0 : 8;
#pragma unroll
        for (int h = 0; h < 16; ++h) {
            int ka = ((h < 8) ? h : h + 8) + kbase;   // ISA 16-bit A 16x32 layout
            a[h] = (_Float16)xr[ka];
        }
        const float* wr = Wt + (size_t)n * CC + k0 + (lo ? 0 : 8);
#pragma unroll
        for (int h = 0; h < 16; ++h) b[h] = (_Float16)wr[(h < 8) ? h : h + 8];
        acc = __builtin_amdgcn_wmma_f32_16x16x32_f16(false, a, false, b,
                                                     (short)0, acc, false, false);
    }
    const float bv = bias ? bias[tn + l15] : 0.0f;
#pragma unroll
    for (int r = 0; r < 8; ++r) {
        int mm = tm + r + (lo ? 0 : 8);
        OUT[(size_t)mm * CC + tn + l15] = acc[r] + bv;
    }
}

// ======================================================================
// build xs2[bk][l][c] from v (4 scan directions)
// ======================================================================
__global__ void k_build_xs(const float* __restrict__ v, float* __restrict__ xs2) {
    size_t idx = (size_t)blockIdx.x * blockDim.x + threadIdx.x;
    const size_t total = (size_t)BKN * LL * CC;
    if (idx >= total) return;
    int bk = (int)(idx / ((size_t)LL * CC));
    int rem = (int)(idx % ((size_t)LL * CC));
    int l = rem / CC, d = rem % CC;
    int b = bk >> 2, kd = bk & 3;
    int lsrc;
    if (kd == 0)      lsrc = l;
    else if (kd == 1) { int w = l >> 6, h = l & 63; lsrc = h * WW + w; }
    else if (kd == 2) lsrc = LL - 1 - l;
    else              { int j = LL - 1 - l; int w = j >> 6, h = j & 63; lsrc = h * WW + w; }
    xs2[idx] = v[((size_t)b * LL + lsrc) * CC + d];
}

// ======================================================================
// x_dbl[bk][l][c'] (c'<48, rows>=38 zero)  -- WMMA f16
// ======================================================================
__global__ void k_xdbl(const float* __restrict__ xs2,
                       const float* __restrict__ xpw,   // (K,38,C)
                       float* __restrict__ xdbl) {
    const int bk = blockIdx.y;
    const int kd = bk & 3;
    const int wave = threadIdx.x >> 5;
    const int lane = threadIdx.x & 31;
    const int tilesN = DBLP / 16;                      // 3
    const int tile = blockIdx.x * 8 + wave;            // 0..767
    const int tm = (tile / tilesN) * 16;
    const int tn = (tile % tilesN) * 16;

    const int l15 = lane & 15;
    const bool lo = (lane < 16);
    const int m = tm + l15;
    const int n = tn + l15;
    const bool nvalid = (n < DBL);

    const float* A0 = xs2 + ((size_t)bk * LL + m) * CC;
    v8f acc = {};
#pragma unroll
    for (int k0 = 0; k0 < CC; k0 += 32) {
        v16h a, b;
        const int kbase = lo ? 0 : 8;
#pragma unroll
        for (int h = 0; h < 16; ++h) {
            int ka = ((h < 8) ? h : h + 8) + kbase;
            a[h] = (_Float16)A0[k0 + ka];
        }
        const float* wr = xpw + ((size_t)kd * DBL + (nvalid ? n : 0)) * CC + k0 + (lo ? 0 : 8);
#pragma unroll
        for (int h = 0; h < 16; ++h)
            b[h] = nvalid ? (_Float16)wr[(h < 8) ? h : h + 8] : (_Float16)0.0f;
        acc = __builtin_amdgcn_wmma_f32_16x16x32_f16(false, a, false, b,
                                                     (short)0, acc, false, false);
    }
#pragma unroll
    for (int r = 0; r < 8; ++r) {
        int mm = tm + r + (lo ? 0 : 8);
        xdbl[((size_t)bk * LL + mm) * DBLP + tn + l15] = acc[r];
    }
}

// ======================================================================
// delta[bk][l][d] = softplus( sum_r xdbl[bk][l][r]*dt_w[kd][d][r] + dt_b )
// ======================================================================
__global__ void k_delta(const float* __restrict__ xdbl,
                        const float* __restrict__ dt_w,   // (K,C,R)
                        const float* __restrict__ dt_b,   // (K,C)
                        float* __restrict__ delta) {
    size_t idx = (size_t)blockIdx.x * blockDim.x + threadIdx.x;
    const size_t total = (size_t)BKN * LL * CC;
    if (idx >= total) return;
    int bk = (int)(idx / ((size_t)LL * CC));
    int rem = (int)(idx % ((size_t)LL * CC));
    int l = rem / CC, d = rem % CC;
    int kd = bk & 3;
    const float* xr = xdbl + ((size_t)bk * LL + l) * DBLP;
    const float* wr = dt_w + ((size_t)kd * CC + d) * RR;
    float s = dt_b[kd * CC + d];
#pragma unroll
    for (int r = 0; r < RR; ++r) s += xr[r] * wr[r];
    delta[idx] = softplusf(s);
}

// ======================================================================
// Chunked selective scan.
// Recurrence per (bk,c,n):  h <- exp(d*A_n)*h + d*u*B_n ;  y_l = sum_n h*C_n
// Phase 1: per chunk, affine composition  (P = prod dA, S = local state)
// Phase 2: sequential over 32 chunks -> true initial state per chunk
// Phase 3: per chunk, re-run from true init, emit y + D*u
// Chunk-state layout: [bk][chunk][c][n]
// ======================================================================
__global__ void k_scan_p1(const float* __restrict__ delta,
                          const float* __restrict__ xs2,
                          const float* __restrict__ xdbl,
                          const float* __restrict__ A_logs,  // (K*C, N)
                          float* __restrict__ Pst, float* __restrict__ Sst) {
    const int j  = blockIdx.x;
    const int bk = blockIdx.y;
    const int kd = bk & 3;
    const int c  = threadIdx.x;
    if (c >= CC) return;

    float Av[NN], pr[NN], st[NN];
#pragma unroll
    for (int n = 0; n < NN; ++n) {
        Av[n] = -__expf(A_logs[((size_t)kd * CC + c) * NN + n]);
        pr[n] = 1.0f;
        st[n] = 0.0f;
    }
    const float* dl = delta + ((size_t)bk * LL + (size_t)j * TCH) * CC;
    const float* ul = xs2   + ((size_t)bk * LL + (size_t)j * TCH) * CC;
    const float* xd = xdbl  + ((size_t)bk * LL + (size_t)j * TCH) * DBLP;

    for (int t = 0; t < TCH; ++t) {
        float d  = dl[(size_t)t * CC + c];
        float du = d * ul[(size_t)t * CC + c];
        const float* Bv = xd + (size_t)t * DBLP + RR;
#pragma unroll
        for (int n = 0; n < NN; ++n) {
            float dA = __expf(d * Av[n]);
            pr[n] *= dA;
            st[n] = dA * st[n] + du * Bv[n];
        }
    }
#pragma unroll
    for (int n = 0; n < NN; ++n) {
        size_t o = (((size_t)bk * NCHUNK + j) * CC + c) * NN + n;
        Pst[o] = pr[n]; Sst[o] = st[n];
    }
}

__global__ void k_scan_p2(const float* __restrict__ Pst,
                          const float* __restrict__ Sst,
                          float* __restrict__ H0) {
    int tid = blockIdx.x * blockDim.x + threadIdx.x;   // bk*CC + c
    if (tid >= BKN * CC) return;
    int bk = tid / CC, c = tid % CC;
    float h[NN];
#pragma unroll
    for (int n = 0; n < NN; ++n) h[n] = 0.0f;
    for (int j = 0; j < NCHUNK; ++j) {
        size_t o = (((size_t)bk * NCHUNK + j) * CC + c) * NN;
#pragma unroll
        for (int n = 0; n < NN; ++n) {
            H0[o + n] = h[n];
            h[n] = Pst[o + n] * h[n] + Sst[o + n];
        }
    }
}

__global__ void k_scan_p3(const float* __restrict__ delta,
                          const float* __restrict__ xs2,
                          const float* __restrict__ xdbl,
                          const float* __restrict__ A_logs,
                          const float* __restrict__ Ds,     // (K*C)
                          const float* __restrict__ H0,
                          float* __restrict__ oy) {
    const int j  = blockIdx.x;
    const int bk = blockIdx.y;
    const int kd = bk & 3;
    const int c  = threadIdx.x;
    if (c >= CC) return;

    float Av[NN], h[NN];
    {
        size_t o = (((size_t)bk * NCHUNK + j) * CC + c) * NN;
#pragma unroll
        for (int n = 0; n < NN; ++n) {
            Av[n] = -__expf(A_logs[((size_t)kd * CC + c) * NN + n]);
            h[n]  = H0[o + n];
        }
    }
    const float Dv = Ds[kd * CC + c];
    const float* dl = delta + ((size_t)bk * LL + (size_t)j * TCH) * CC;
    const float* ul = xs2   + ((size_t)bk * LL + (size_t)j * TCH) * CC;
    const float* xd = xdbl  + ((size_t)bk * LL + (size_t)j * TCH) * DBLP;
    float* out = oy + ((size_t)bk * LL + (size_t)j * TCH) * CC;

    for (int t = 0; t < TCH; ++t) {
        float d  = dl[(size_t)t * CC + c];
        float u  = ul[(size_t)t * CC + c];
        float du = d * u;
        const float* Bv = xd + (size_t)t * DBLP + RR;
        const float* Cv = Bv + NN;
        float y = 0.0f;
#pragma unroll
        for (int n = 0; n < NN; ++n) {
            h[n] = __expf(d * Av[n]) * h[n] + du * Bv[n];
            y += h[n] * Cv[n];
        }
        out[(size_t)t * CC + c] = y + Dv * u;
    }
}

// ======================================================================
// merge 4 directions -> att[b][l][c]
// ======================================================================
__global__ void k_merge(const float* __restrict__ oy, float* __restrict__ att) {
    size_t idx = (size_t)blockIdx.x * blockDim.x + threadIdx.x;
    const size_t total = (size_t)BSZ * LL * CC;
    if (idx >= total) return;
    int b = (int)(idx / ((size_t)LL * CC));
    int rem = (int)(idx % ((size_t)LL * CC));
    int l = rem / CC, c = rem % CC;
    int h = l >> 6, w = l & 63;
    int p = w * HH + h;
    size_t base = (size_t)b * 4 * LL * CC;
    float t = oy[base + (size_t)(0 * LL + l)            * CC + c]
            + oy[base + (size_t)(2 * LL + (LL - 1 - l)) * CC + c]
            + oy[base + (size_t)(1 * LL + p)            * CC + c]
            + oy[base + (size_t)(3 * LL + (LL - 1 - p)) * CC + c];
    att[idx] = t;
}

// ======================================================================
// depthwise 3x3 conv + BN + GELU  -> convx[b][l][c]
// ======================================================================
__global__ void k_dwconv(const float* __restrict__ v,
                         const float* __restrict__ dw_w,  // (C,1,3,3)
                         const float* __restrict__ dw_b,
                         const float* __restrict__ g,
                         const float* __restrict__ bb,
                         float* __restrict__ convx) {
    size_t idx = (size_t)blockIdx.x * blockDim.x + threadIdx.x;
    const size_t total = (size_t)BSZ * LL * CC;
    if (idx >= total) return;
    int b = (int)(idx / ((size_t)LL * CC));
    int rem = (int)(idx % ((size_t)LL * CC));
    int l = rem / CC, c = rem % CC;
    int h = l >> 6, w = l & 63;
    float s = 0.0f;
#pragma unroll
    for (int dh = -1; dh <= 1; ++dh) {
#pragma unroll
        for (int dw = -1; dw <= 1; ++dw) {
            int hh2 = h + dh, ww2 = w + dw;
            if (hh2 >= 0 && hh2 < HH && ww2 >= 0 && ww2 < WW) {
                s += v[((size_t)b * LL + hh2 * WW + ww2) * CC + c]
                   * dw_w[c * 9 + (dh + 1) * 3 + (dw + 1)];
            }
        }
    }
    s += dw_b[c];
    s = s * (g[c] * BNSCALE) + bb[c];
    convx[idx] = gelu_exact(s);
}

// ======================================================================
// global pool + channel-interaction MLP -> sigc[b][c]
// ======================================================================
__global__ void k_pool_cm(const float* __restrict__ convx,
                          const float* __restrict__ ci_w1, const float* __restrict__ ci_b1,
                          const float* __restrict__ ci_g,  const float* __restrict__ ci_bb,
                          const float* __restrict__ ci_w2, const float* __restrict__ ci_b2,
                          float* __restrict__ sigc) {
    __shared__ float pooled[CC];
    __shared__ float hid[CC / 8];
    int b = blockIdx.x, t = threadIdx.x;
    if (t < CC) {
        float s = 0.0f;
        const float* p = convx + (size_t)b * LL * CC + t;
        for (int l = 0; l < LL; ++l) s += p[(size_t)l * CC];
        pooled[t] = s * (1.0f / (float)LL);
    }
    __syncthreads();
    if (t < CC / 8) {
        float s = ci_b1[t];
        for (int i = 0; i < CC; ++i) s += ci_w1[t * CC + i] * pooled[i];
        s = s * (ci_g[t] * BNSCALE) + ci_bb[t];
        hid[t] = gelu_exact(s);
    }
    __syncthreads();
    if (t < CC) {
        float s = ci_b2[t];
#pragma unroll
        for (int j = 0; j < CC / 8; ++j) s += ci_w2[t * (CC / 8) + j] * hid[j];
        sigc[b * CC + t] = sigmoidf(s);
    }
}

// ======================================================================
// spatial-interaction MLP -> sigs[b][l]
// ======================================================================
__global__ void k_spatial(const float* __restrict__ att,
                          const float* __restrict__ si_w1, const float* __restrict__ si_b1,
                          const float* __restrict__ si_g,  const float* __restrict__ si_bb,
                          const float* __restrict__ si_w2, const float* __restrict__ si_b2,
                          float* __restrict__ sigs) {
    int idx = blockIdx.x * blockDim.x + threadIdx.x;
    if (idx >= BSZ * LL) return;
    const float* row = att + (size_t)idx * CC;
    float sm = si_b2[0];
#pragma unroll
    for (int j = 0; j < CC / 16; ++j) {       // 6 hidden
        float s = si_b1[j];
        for (int c = 0; c < CC; ++c) s += si_w1[j * CC + c] * row[c];
        s = s * (si_g[j] * BNSCALE) + si_bb[j];
        sm += si_w2[j] * gelu_exact(s);
    }
    sigs[idx] = sigmoidf(sm);
}

// ======================================================================
// fused = att*sig(channel) + convx*sig(spatial)
// ======================================================================
__global__ void k_fuse(const float* __restrict__ att, const float* __restrict__ convx,
                       const float* __restrict__ sigc, const float* __restrict__ sigs,
                       float* __restrict__ fused) {
    size_t idx = (size_t)blockIdx.x * blockDim.x + threadIdx.x;
    const size_t total = (size_t)BSZ * LL * CC;
    if (idx >= total) return;
    int b = (int)(idx / ((size_t)LL * CC));
    int rem = (int)(idx % ((size_t)LL * CC));
    int l = rem / CC, c = rem % CC;
    fused[idx] = att[idx] * sigc[b * CC + c] + convx[idx] * sigs[b * LL + l];
}

// ======================================================================
extern "C" void kernel_launch(void* const* d_in, const int* in_sizes, int n_in,
                              void* d_out, int out_size, void* d_ws, size_t ws_size,
                              hipStream_t stream) {
    (void)in_sizes; (void)n_in; (void)out_size; (void)ws_size;
    const float* x      = (const float*)d_in[0];
    const float* qkv_w  = (const float*)d_in[3];
    const float* proj_w = (const float*)d_in[4];
    const float* proj_b = (const float*)d_in[5];
    const float* dw_w   = (const float*)d_in[6];
    const float* dw_b   = (const float*)d_in[7];
    const float* bn1_g  = (const float*)d_in[8];
    const float* bn1_b  = (const float*)d_in[9];
    const float* ci_w1  = (const float*)d_in[10];
    const float* ci_b1  = (const float*)d_in[11];
    const float* ci_g   = (const float*)d_in[12];
    const float* ci_bb  = (const float*)d_in[13];
    const float* ci_w2  = (const float*)d_in[14];
    const float* ci_b2  = (const float*)d_in[15];
    const float* si_w1  = (const float*)d_in[16];
    const float* si_b1  = (const float*)d_in[17];
    const float* si_g   = (const float*)d_in[18];
    const float* si_bb  = (const float*)d_in[19];
    const float* si_w2  = (const float*)d_in[20];
    const float* si_b2  = (const float*)d_in[21];
    const float* xpw    = (const float*)d_in[22];
    const float* dt_w   = (const float*)d_in[23];
    const float* dt_b   = (const float*)d_in[24];
    const float* A_logs = (const float*)d_in[25];
    const float* Ds     = (const float*)d_in[26];

    float* ws = (float*)d_ws;
    const size_t SZ_BLC  = (size_t)BSZ * LL * CC;            //   786432
    const size_t SZ_BKLC = (size_t)BKN * LL * CC;            //  3145728
    const size_t SZ_XDBL = (size_t)BKN * LL * DBLP;          //  1572864
    const size_t SZ_CHNK = (size_t)BKN * NCHUNK * CC * NN;   //   393216
    float* v     = ws;                       size_t off = SZ_BLC;
    float* xs2   = ws + off;                 off += SZ_BKLC;
    float* xdbl  = ws + off;                 off += SZ_XDBL;
    float* delta = ws + off;                 off += SZ_BKLC;
    float* oy    = ws + off;                 off += SZ_BKLC;
    float* att   = ws + off;                 off += SZ_BLC;
    float* convx = ws + off;                 off += SZ_BLC;
    float* sigc  = ws + off;                 off += (size_t)BSZ * CC;
    float* sigs  = ws + off;                 off += (size_t)BSZ * LL;
    float* fused = ws + off;                 off += SZ_BLC;
    float* Pst   = ws + off;                 off += SZ_CHNK;
    float* Sst   = ws + off;                 off += SZ_CHNK;
    float* H0    = ws + off;                 off += SZ_CHNK;

    const int M = BSZ * LL;                                  // 8192

    // 1. qkv GEMM (WMMA)
    k_wmma_gemm_nt<<<(M / 16) * (CC / 16) / 8, 256, 0, stream>>>(x, qkv_w, nullptr, v, M);

    // 2. direction gather
    k_build_xs<<<(unsigned)((SZ_BKLC + 255) / 256), 256, 0, stream>>>(v, xs2);

    // 3. x_dbl GEMM (WMMA)
    {
        dim3 grid((LL / 16) * (DBLP / 16) / 8, BKN);
        k_xdbl<<<grid, 256, 0, stream>>>(xs2, xpw, xdbl);
    }

    // 4. delta (softplus of dt projection)
    k_delta<<<(unsigned)((SZ_BKLC + 255) / 256), 256, 0, stream>>>(xdbl, dt_w, dt_b, delta);

    // 5. chunked selective scan: 32x parallel chunks + tiny sequential fixup
    {
        dim3 gchunk(NCHUNK, BKN);
        k_scan_p1<<<gchunk, CC, 0, stream>>>(delta, xs2, xdbl, A_logs, Pst, Sst);
        k_scan_p2<<<(BKN * CC + 95) / 96, 96, 0, stream>>>(Pst, Sst, H0);
        k_scan_p3<<<gchunk, CC, 0, stream>>>(delta, xs2, xdbl, A_logs, Ds, H0, oy);
    }

    // 6. merge directions
    k_merge<<<(unsigned)((SZ_BLC + 255) / 256), 256, 0, stream>>>(oy, att);

    // 7. depthwise conv + BN + GELU
    k_dwconv<<<(unsigned)((SZ_BLC + 255) / 256), 256, 0, stream>>>(v, dw_w, dw_b, bn1_g, bn1_b, convx);

    // 8. channel interaction
    k_pool_cm<<<BSZ, 128, 0, stream>>>(convx, ci_w1, ci_b1, ci_g, ci_bb, ci_w2, ci_b2, sigc);

    // 9. spatial interaction
    k_spatial<<<(BSZ * LL + 255) / 256, 256, 0, stream>>>(att, si_w1, si_b1, si_g, si_bb, si_w2, si_b2, sigs);

    // 10. gated fuse
    k_fuse<<<(unsigned)((SZ_BLC + 255) / 256), 256, 0, stream>>>(att, convx, sigc, sigs, fused);

    // 11. output projection (WMMA) + bias -> d_out
    k_wmma_gemm_nt<<<(M / 16) * (CC / 16) / 8, 256, 0, stream>>>(fused, proj_w, proj_b, (float*)d_out, M);
}
